// Interaction_Block_28217935135448
// MI455X (gfx1250) — matrix operand order repaired and checked by reading, not compiled, not run on hardware
//
#include <hip/hip_runtime.h>
#include <math.h>

// ====================================================================
// gfx1250 wave32 WMMA helpers (v_wmma_f32_16x16x32_f16)
// ====================================================================

typedef _Float16 f16;
typedef __attribute__((ext_vector_type(16))) _Float16 v16h;
typedef __attribute__((ext_vector_type(8)))  _Float16 v8h;
typedef __attribute__((ext_vector_type(8)))  float    v8f;

union AFrag { v16h v; v8h p[2]; f16 h[16]; };
union CFrag { v8f  v; float f[8]; };

// 32B-granule XOR swizzle: avoids LDS bank conflicts on row-strided
// fragment loads.  ld (in halves) must be a multiple of 128.
__device__ __forceinline__ int swz(int row, int col, int ld) {
  return row * ld + ((((col >> 4) ^ (row & 7)) << 4) | (col & 15));
}

__device__ __forceinline__ void zero_acc(CFrag acc[8]) {
#pragma unroll
  for (int t = 0; t < 8; ++t) {
#pragma unroll
    for (int j = 0; j < 8; ++j) acc[t].f[j] = 0.f;
  }
}

// acc[16 rows m0.. x 128 cols] += A[rows, colbase..colbase+128) * W
// Wt holds W transposed (logical Wt[n][k]) swizzled, one 128-K chunk.
__device__ __forceinline__ void wave_gemm128(CFrag acc[8],
    const f16* __restrict__ A, int lda, int colbase,
    const f16* __restrict__ Wt, int m0, int lane)
{
  const int g   = lane >> 4;      // lane group (0/1)
  const int mr  = lane & 15;
  const int row = m0 + mr;
  const int rx  = row & 7;
  const f16* arow = A + row * lda;
#pragma unroll
  for (int k0 = 0; k0 < 128; k0 += 32) {
    const int c0 = (colbase + k0) >> 4;
    // A 16x32 f16 fragment: lanes 0-15 hold K k0..k0+15,
    // lanes 16-31 hold K k0+16..31.
    AFrag a;
    a.p[0] = *(const v8h*)(arow + (((c0 + g) ^ rx) << 4));
    a.p[1] = *(const v8h*)(arow + (((c0 + g) ^ rx) << 4) + 8);
#pragma unroll
    for (int t = 0; t < 8; ++t) {
      const int n  = t * 16 + mr;       // output column
      const int cb = (k0 >> 4) + g;     // 16-half K-chunk within Wt row
      AFrag b;
      b.v = *(const v16h*)(Wt + n * 128 + ((cb ^ (n & 7)) << 4));
      acc[t].v = __builtin_amdgcn_wmma_f32_16x16x32_f16(
          false, a.v, false, b.v, (short)0, acc[t].v, false, false);
    }
  }
}

__device__ __forceinline__ void store_acc(const CFrag acc[8], float* Y,
                                          int m0, int lane)
{
  const int g = lane >> 4, mr = lane & 15;
#pragma unroll
  for (int t = 0; t < 8; ++t) {
#pragma unroll
    for (int j = 0; j < 8; ++j)
      Y[(m0 + j + 8 * g) * 128 + t * 16 + mr] = acc[t].f[j];
  }
}

// Load one 128x128 f32 weight block (row-major [k][j]) transposed into
// swizzled f16 LDS: Wt[n=j][k].
__device__ __forceinline__ void load_w_t(const float* __restrict__ Wg,
                                         f16* Wt, int tid)
{
  for (int idx = tid; idx < 128 * 128; idx += 256) {
    const int k = idx & 127, j = idx >> 7;
    Wt[swz(j, k, 128)] = (f16)Wg[k * 128 + j];
  }
}

// y[128] (+b2) -> LayerNorm(g,beta) -> f16 into swizzled Aact column range
__device__ __forceinline__ void ln_to_f16(const float* __restrict__ y,
    const float* __restrict__ b2, const float* __restrict__ gam,
    const float* __restrict__ bet, f16* Abase, int row, int colbase,
    bool accum)
{
  float s = 0.f, s2 = 0.f;
  for (int c = 0; c < 128; ++c) { float v = y[c] + b2[c]; s += v; s2 += v * v; }
  const float mu = s * (1.f / 128.f);
  const float rs = rsqrtf(fmaxf(s2 * (1.f / 128.f) - mu * mu, 0.f) + 1e-5f);
  for (int c = 0; c < 128; ++c) {
    float v = (y[c] + b2[c] - mu) * rs * gam[c] + bet[c];
    const int id = swz(row, colbase + c, 384);
    if (accum) v += (float)Abase[id];
    Abase[id] = (f16)v;
  }
}

// ====================================================================
// Node MLPs: nw / mi / ii / de   (128 -> relu 128 -> 1)
// ====================================================================

struct NodeParams {
  const float* nlat;
  const float* W1[4]; const float* b1[4];
  const float* W2[4]; const float* b2[4];
  float* out[4];
  int N;
};

#define NODE_SMEM (32768 + 32768 + 65536)

extern "C" __global__ __launch_bounds__(256)
void ib_node_kernel(NodeParams p)
{
  extern __shared__ char smem[];
  f16*   A    = (f16*)(smem);             // 128x128 f16 (swizzled)
  f16*   Wt   = (f16*)(smem + 32768);     // 128x128 f16 (transposed+swz)
  float* Ybuf = (float*)(smem + 65536);   // 128x128 f32

  const int tid  = threadIdx.x;
  const int lane = tid & 31;
  const int m0   = (tid >> 5) * 16;
  const int n0   = blockIdx.x * 128;

  for (int idx = tid; idx < 128 * 128; idx += 256) {
    const int r = idx >> 7, c = idx & 127;
    const int n = n0 + r;
    A[swz(r, c, 128)] = (n < p.N) ? (f16)p.nlat[(size_t)n * 128 + c] : (f16)0.f;
  }
  for (int m = 0; m < 4; ++m) {
    __syncthreads();
    load_w_t(p.W1[m], Wt, tid);
    __syncthreads();
    CFrag acc[8]; zero_acc(acc);
    wave_gemm128(acc, A, 128, 0, Wt, m0, lane);
    store_acc(acc, Ybuf, m0, lane);
    __syncthreads();
    if (tid < 128) {
      const int n = n0 + tid;
      if (n < p.N) {
        float o = p.b2[m][0];
        const float* y = Ybuf + tid * 128;
        for (int c = 0; c < 128; ++c)
          o += fmaxf(y[c] + p.b1[m][c], 0.f) * p.W2[m][c];
        p.out[m][n] = o;
      }
    }
  }
}

// ====================================================================
// Fused edge pipeline
// ====================================================================

struct EdgeParams {
  const int* ei;                                     // [2E]
  const float *spos, *rpos, *dx, *dt, *va, *vb, *vc;
  const float *svt, *svtm, *swt, *rvt, *rvtm, *rwt;
  const float *eattr, *nlat;
  const float *eeW1, *eeb1, *eeW2, *eeb2, *eeg, *eebt;
  const float *efW1, *efb1, *efW2, *efb2, *efg, *efbt;
  const float *ieW1, *ieb1, *ieW2, *ieb2, *ieg, *iebt;
  const float *i1W1, *i1b1, *i1W2, *i1b2;
  const float *i2W1, *i2b1, *i2W2, *i2b2;
  const float *fsW1, *fsb1, *fsW2, *fsb2;
  const float *nw;                                   // [N] node weights
  float *accF, *accT;                                // [N*3] scatter accums
  float *out_inter;                                  // d_out + 6N, [E*128]
  int E, N;
};

// LDS: Wt 32KB | Aact 128x384 f16 96KB | Hbuf 128x128 f16 32KB
//    | Ybuf 128x128 f32 64KB | SF 128x24 f32 12KB  => 236KB dynamic
#define EDGE_SMEM (32768 + 98304 + 32768 + 65536 + 12288)

extern "C" __global__ __launch_bounds__(256)
void ib_edge_kernel(EdgeParams p)
{
  extern __shared__ char smem[];
  f16*   Wt   = (f16*)(smem);
  f16*   Aact = (f16*)(smem + 32768);
  f16*   Hbuf = (f16*)(smem + 131072);
  float* Ybuf = (float*)(smem + 163840);
  float* SF   = (float*)(smem + 229376);

  const int tid  = threadIdx.x;
  const int lane = tid & 31;
  const int m0   = (tid >> 5) * 16;
  const int e0   = blockIdx.x * 128;

  // ---- edge scalar features: sf[9], rf[9] (negated), edge_in[3] ----
  if (tid < 128) {
    const int e = e0 + tid;
    float* sf = SF + tid * 24;
    if (e < p.E) {
      const float ax = p.va[3*e], ay = p.va[3*e+1], az = p.va[3*e+2];
      const float bx = p.vb[3*e], by = p.vb[3*e+1], bz = p.vb[3*e+2];
      const float cx = p.vc[3*e], cy = p.vc[3*e+1], cz = p.vc[3*e+2];
      const float* srcs[3] = { p.svt, p.svtm, p.swt };
      const float* srcr[3] = { p.rvt, p.rvtm, p.rwt };
      for (int q = 0; q < 3; ++q) {
        float vx = srcs[q][3*e], vy = srcs[q][3*e+1], vz = srcs[q][3*e+2];
        sf[q*3+0] = vx*ax + vy*ay + vz*az;
        sf[q*3+1] = vx*bx + vy*by + vz*bz;
        sf[q*3+2] = vx*cx + vy*cy + vz*cz;
      }
      for (int q = 0; q < 3; ++q) {
        float vx = srcr[q][3*e], vy = srcr[q][3*e+1], vz = srcr[q][3*e+2];
        sf[9+q*3+0] = -(vx*ax + vy*ay + vz*az);
        sf[9+q*3+1] = -(vx*bx + vy*by + vz*bz);
        sf[9+q*3+2] = -(vx*cx + vy*cy + vz*cz);
      }
      float x0 = p.dx[3*e], x1 = p.dx[3*e+1], x2 = p.dx[3*e+2];
      float t0 = p.dt[3*e], t1 = p.dt[3*e+1], t2 = p.dt[3*e+2];
      sf[18] = sqrtf(x0*x0 + x1*x1 + x2*x2);
      sf[19] = sqrtf(t0*t0 + t1*t1 + t2*t2);
      sf[20] = p.eattr[e];
    } else {
      for (int q = 0; q < 21; ++q) sf[q] = 0.f;
    }
  }
  __syncthreads();

  // ---- node_latent[s] + node_latent[r]  ->  Aact[:,128:256] ----
  for (int idx = tid; idx < 128 * 128; idx += 256) {
    const int er = idx >> 7, c = idx & 127;
    const int e = e0 + er;
    float v = 0.f;
    if (e < p.E) {
      const int s = p.ei[e], r = p.ei[p.E + e];
      v = p.nlat[(size_t)s * 128 + c] + p.nlat[(size_t)r * 128 + c];
    }
    Aact[swz(er, 128 + c, 384)] = (f16)v;
  }

  // ---- ee: edge_in(3) -> relu 128 (VALU) -> 128 (WMMA) -> LN ----
  for (int idx = tid; idx < 128 * 128; idx += 256) {
    const int er = idx >> 7, c = idx & 127;
    const float* s = SF + er * 24 + 18;
    float h = p.eeb1[c] + s[0]*p.eeW1[c] + s[1]*p.eeW1[128+c] + s[2]*p.eeW1[256+c];
    Hbuf[swz(er, c, 128)] = (f16)fmaxf(h, 0.f);
  }
  __syncthreads();
  load_w_t(p.eeW2, Wt, tid);
  __syncthreads();
  { CFrag acc[8]; zero_acc(acc);
    wave_gemm128(acc, Hbuf, 128, 0, Wt, m0, lane);
    store_acc(acc, Ybuf, m0, lane); }
  __syncthreads();
  if (tid < 128)
    ln_to_f16(Ybuf + tid * 128, p.eeb2, p.eeg, p.eebt, Aact, tid, 256, false);
  __syncthreads();

  // ---- ef applied to senders then receivers, summed into Aact[:,0:128] ----
  load_w_t(p.efW2, Wt, tid);
  __syncthreads();
  for (int sp = 0; sp < 2; ++sp) {
    for (int idx = tid; idx < 128 * 128; idx += 256) {
      const int er = idx >> 7, c = idx & 127;
      const float* s = SF + er * 24 + sp * 9;
      float h = p.efb1[c];
#pragma unroll
      for (int k = 0; k < 9; ++k) h += s[k] * p.efW1[k * 128 + c];
      Hbuf[swz(er, c, 128)] = (f16)fmaxf(h, 0.f);
    }
    __syncthreads();
    { CFrag acc[8]; zero_acc(acc);
      wave_gemm128(acc, Hbuf, 128, 0, Wt, m0, lane);
      store_acc(acc, Ybuf, m0, lane); }
    __syncthreads();
    if (tid < 128)
      ln_to_f16(Ybuf + tid * 128, p.efb2, p.efg, p.efbt, Aact, tid, 0, sp == 1);
    __syncthreads();
  }

  // ---- ie layer1: [128x384] @ [384x128], K chunked over 3 weight tiles ----
  {
    CFrag acc[8]; zero_acc(acc);
    for (int ch = 0; ch < 3; ++ch) {
      __syncthreads();
      load_w_t(p.ieW1 + (size_t)ch * 128 * 128, Wt, tid);
      __syncthreads();
      wave_gemm128(acc, Aact, 384, ch * 128, Wt, m0, lane);
    }
    store_acc(acc, Ybuf, m0, lane);
  }
  __syncthreads();
  for (int idx = tid; idx < 128 * 128; idx += 256) {
    const int er = idx >> 7, c = idx & 127;
    Hbuf[swz(er, c, 128)] = (f16)fmaxf(Ybuf[idx] + p.ieb1[c], 0.f);
  }
  __syncthreads();

  // ---- ie layer2 + LN: interaction_latent -> d_out (f32) + Aact[:,0:128] ----
  load_w_t(p.ieW2, Wt, tid);
  __syncthreads();
  { CFrag acc[8]; zero_acc(acc);
    wave_gemm128(acc, Hbuf, 128, 0, Wt, m0, lane);
    store_acc(acc, Ybuf, m0, lane); }
  __syncthreads();
  if (tid < 128) {
    const int e = e0 + tid;
    const float* y = Ybuf + tid * 128;
    float s = 0.f, s2 = 0.f;
    for (int c = 0; c < 128; ++c) { float v = y[c] + p.ieb2[c]; s += v; s2 += v*v; }
    const float mu = s * (1.f / 128.f);
    const float rs = rsqrtf(fmaxf(s2 * (1.f / 128.f) - mu * mu, 0.f) + 1e-5f);
    for (int c = 0; c < 128; ++c) {
      float v = (y[c] + p.ieb2[c] - mu) * rs * p.ieg[c] + p.iebt[c];
      if (e < p.E) p.out_inter[(size_t)e * 128 + c] = v;
      Aact[swz(tid, c, 384)] = (f16)v;
    }
  }
  __syncthreads();

  // ---- decoders i1 / i2 / fs: 128 -> relu 128 (WMMA) -> {3,3,1} (VALU) ----
  {
    const float* W1s[3] = { p.i1W1, p.i2W1, p.fsW1 };
    const float* b1s[3] = { p.i1b1, p.i2b1, p.fsb1 };
    const float* W2s[3] = { p.i1W2, p.i2W2, p.fsW2 };
    const float* b2s[3] = { p.i1b2, p.i2b2, p.fsb2 };
    for (int d = 0; d < 3; ++d) {
      load_w_t(W1s[d], Wt, tid);
      __syncthreads();
      { CFrag acc[8]; zero_acc(acc);
        wave_gemm128(acc, Aact, 384, 0, Wt, m0, lane);
        store_acc(acc, Ybuf, m0, lane); }
      __syncthreads();
      if (tid < 128) {
        const int dn = (d < 2) ? 3 : 1;
        float o0 = b2s[d][0];
        float o1 = (dn == 3) ? b2s[d][1] : 0.f;
        float o2 = (dn == 3) ? b2s[d][2] : 0.f;
        const float* y = Ybuf + tid * 128;
        for (int c = 0; c < 128; ++c) {
          const float h = fmaxf(y[c] + b1s[d][c], 0.f);
          o0 += h * W2s[d][c * dn];
          if (dn == 3) { o1 += h * W2s[d][c * 3 + 1]; o2 += h * W2s[d][c * 3 + 2]; }
        }
        float* o = SF + tid * 24 + d * 3;   // cf[0..2], ca[3..5], lam[6]
        o[0] = o0;
        if (dn == 3) { o[1] = o1; o[2] = o2; }
      }
      __syncthreads();
    }
  }

  // ---- epilogue: fij / tauij, scatter-add onto receivers ----
  if (tid < 128) {
    const int e = e0 + tid;
    if (e < p.E) {
      const int s = p.ei[e], r = p.ei[p.E + e];
      const float wsn = p.nw[s], wrn = p.nw[r];
      const float inv = 1.f / (wsn + wrn);
      const float ax = p.va[3*e], ay = p.va[3*e+1], az = p.va[3*e+2];
      const float bx = p.vb[3*e], by = p.vb[3*e+1], bz = p.vb[3*e+2];
      const float cx = p.vc[3*e], cy = p.vc[3*e+1], cz = p.vc[3*e+2];
      const float* o = SF + tid * 24;
      const float cf0 = o[0], cf1 = o[1], cf2 = o[2];
      const float ca0 = o[3], ca1 = o[4], ca2 = o[5];
      const float lam = o[6];
      const float fx = cf0*ax + cf1*bx + cf2*cx;
      const float fy = cf0*ay + cf1*by + cf2*cy;
      const float fz = cf0*az + cf1*bz + cf2*cz;
      const float gx = ca0*ax + ca1*bx + ca2*cx;
      const float gy = ca0*ay + ca1*by + ca2*cy;
      const float gz = ca0*az + ca1*bz + ca2*cz;
      const float rx = p.rpos[3*e], ry = p.rpos[3*e+1], rz = p.rpos[3*e+2];
      const float sx = p.spos[3*e], sy = p.spos[3*e+1], sz = p.spos[3*e+2];
      const float dxp = rx - (wsn*sx + wrn*rx) * inv;
      const float dyp = ry - (wsn*sy + wrn*ry) * inv;
      const float dzp = rz - (wsn*sz + wrn*rz) * inv;
      const float tx = gx - (dyp*fz - dzp*fy) * lam;
      const float ty = gy - (dzp*fx - dxp*fz) * lam;
      const float tz = gz - (dxp*fy - dyp*fx) * lam;
      atomicAdd(p.accF + 3*r + 0, fx);
      atomicAdd(p.accF + 3*r + 1, fy);
      atomicAdd(p.accF + 3*r + 2, fz);
      atomicAdd(p.accT + 3*r + 0, tx);
      atomicAdd(p.accT + 3*r + 1, ty);
      atomicAdd(p.accT + 3*r + 2, tz);
    }
  }
}

// ====================================================================
// small kernels
// ====================================================================

extern "C" __global__ void ib_zero_kernel(float* p, int n)
{
  const int i = blockIdx.x * blockDim.x + threadIdx.x;
  if (i < n) p[i] = 0.f;
}

extern "C" __global__ void ib_finish_kernel(const float* mi, const float* ii,
    const float* de, const float* accF, const float* accT,
    float* dv, float* dw, int N)
{
  const int i = blockIdx.x * blockDim.x + threadIdx.x;
  if (i < N) {
    const float m = mi[i], iv = ii[i], d = de[i];
#pragma unroll
    for (int k = 0; k < 3; ++k) {
      dv[3*i + k] = m * accF[3*i + k] + d;
      dw[3*i + k] = iv * accT[3*i + k];
    }
  }
}

// ====================================================================
// host launcher
// ====================================================================

#define FIN(i) ((const float*)d_in[i])

extern "C" void kernel_launch(void* const* d_in, const int* in_sizes, int n_in,
                              void* d_out, int out_size, void* d_ws, size_t ws_size,
                              hipStream_t stream)
{
  (void)n_in; (void)out_size; (void)ws_size;
  const int E = in_sizes[1] / 3;
  const int N = in_sizes[15] / 128;

  float* ws   = (float*)d_ws;
  float* nwA  = ws;
  float* miA  = ws + (size_t)N;
  float* iiA  = ws + (size_t)2 * N;
  float* deA  = ws + (size_t)3 * N;
  float* accF = ws + (size_t)4 * N;
  float* accT = ws + (size_t)7 * N;
  float* outF = (float*)d_out;

  (void)hipFuncSetAttribute(reinterpret_cast<const void*>(ib_edge_kernel),
                            hipFuncAttributeMaxDynamicSharedMemorySize, EDGE_SMEM);
  (void)hipFuncSetAttribute(reinterpret_cast<const void*>(ib_node_kernel),
                            hipFuncAttributeMaxDynamicSharedMemorySize, NODE_SMEM);

  // node MLPs: nw(46) mi(50) ii(54) de(58)
  NodeParams np;
  np.nlat = FIN(15);
  for (int m = 0; m < 4; ++m) {
    const int b = 46 + 4 * m;
    np.W1[m] = FIN(b + 0); np.b1[m] = FIN(b + 1);
    np.W2[m] = FIN(b + 2); np.b2[m] = FIN(b + 3);
  }
  np.out[0] = nwA; np.out[1] = miA; np.out[2] = iiA; np.out[3] = deA;
  np.N = N;
  ib_node_kernel<<<(N + 127) / 128, 256, NODE_SMEM, stream>>>(np);

  ib_zero_kernel<<<(6 * N + 255) / 256, 256, 0, stream>>>(accF, 6 * N);

  EdgeParams ep;
  ep.ei    = (const int*)d_in[0];
  ep.spos  = FIN(1);  ep.rpos = FIN(2);  ep.dx = FIN(3);  ep.dt = FIN(4);
  ep.va    = FIN(5);  ep.vb   = FIN(6);  ep.vc = FIN(7);
  ep.svt   = FIN(8);  ep.svtm = FIN(9);  ep.swt = FIN(10);
  ep.rvt   = FIN(11); ep.rvtm = FIN(12); ep.rwt = FIN(13);
  ep.eattr = FIN(14); ep.nlat = FIN(15);
  ep.efW1 = FIN(16); ep.efb1 = FIN(17); ep.efW2 = FIN(18);
  ep.efb2 = FIN(19); ep.efg  = FIN(20); ep.efbt = FIN(21);
  ep.eeW1 = FIN(22); ep.eeb1 = FIN(23); ep.eeW2 = FIN(24);
  ep.eeb2 = FIN(25); ep.eeg  = FIN(26); ep.eebt = FIN(27);
  ep.ieW1 = FIN(28); ep.ieb1 = FIN(29); ep.ieW2 = FIN(30);
  ep.ieb2 = FIN(31); ep.ieg  = FIN(32); ep.iebt = FIN(33);
  ep.i1W1 = FIN(34); ep.i1b1 = FIN(35); ep.i1W2 = FIN(36); ep.i1b2 = FIN(37);
  ep.i2W1 = FIN(38); ep.i2b1 = FIN(39); ep.i2W2 = FIN(40); ep.i2b2 = FIN(41);
  ep.fsW1 = FIN(42); ep.fsb1 = FIN(43); ep.fsW2 = FIN(44); ep.fsb2 = FIN(45);
  ep.nw = nwA; ep.accF = accF; ep.accT = accT;
  ep.out_inter = outF + (size_t)6 * N;
  ep.E = E; ep.N = N;
  ib_edge_kernel<<<(E + 127) / 128, 256, EDGE_SMEM, stream>>>(ep);

  ib_finish_kernel<<<(N + 255) / 256, 256, 0, stream>>>(
      miA, iiA, deA, accF, accT, outF, outF + (size_t)3 * N, N);
}
